// SpikingRNNBase_52991306498505
// MI455X (gfx1250) — matrix operand order repaired and checked by reading, not compiled, not run on hardware
//
#include <hip/hip_runtime.h>

// ---------------------------------------------------------------------------
// Spiking LSTM (heaviside gates) forward for MI455X / gfx1250.
// T=512, B=64, I=H=512, 4H=2048, L=2.
//
// Persistent-kernel scan: 32 workgroups x 512 threads (16 wave32s). Each wave
// owns one 16x16 gate tile per timestep and runs v_wmma_f32_16x16x32_bf16
// over K=512 (input proj) + K=512 (recurrent proj).
//
// v2: weight (B-operand) tiles for this WG are staged into LDS once per layer
// (128 KB, pre-swizzled to the WMMA B lane layout) so the inner t-loop feeds
// WMMA from ds_load_b128 instead of the v1 scratch spills; bias is folded into
// accumulator init; a per-step compiler memory barrier stops LLVM from
// re-hoisting the invariant LDS loads into (spilled) VGPRs.
// ---------------------------------------------------------------------------

typedef __attribute__((ext_vector_type(16))) __bf16 v16bf;
typedef __attribute__((ext_vector_type(8)))  __bf16 v8bf;
typedef __attribute__((ext_vector_type(8)))  float  v8f;

namespace {
constexpr int    T       = 512;
constexpr int    BB      = 64;     // batch
constexpr int    I       = 512;    // input dim
constexpr int    H       = 512;    // hidden dim
constexpr int    G       = 4 * H;  // 2048 gate columns
constexpr int    L       = 2;      // layers
constexpr int    NWG     = 32;     // = H/16 column groups -> grid size
constexpr int    THREADS = 512;    // 16 wave32s per WG
constexpr int    KSTEPS  = I / 32; // 16 wmma k-steps per projection
constexpr size_t YELEMS  = (size_t)T * BB * H; // fp32 y elements in d_out
constexpr size_t BH      = (size_t)BB * H;
}

__device__ __forceinline__ v16bf cat8(v8bf lo, v8bf hi) {
    return __builtin_shufflevector(lo, hi,
        0, 1, 2, 3, 4, 5, 6, 7, 8, 9, 10, 11, 12, 13, 14, 15);
}

// Device-wide barrier: monotonically increasing counter, zeroed each launch.
__device__ __forceinline__ void gbar(unsigned* cnt, unsigned& phase) {
    ++phase;                       // uniform across all threads
    __threadfence();               // release our global stores
    __syncthreads();
    if (threadIdx.x == 0) {
        __hip_atomic_fetch_add(cnt, 1u, __ATOMIC_RELEASE,
                               __HIP_MEMORY_SCOPE_AGENT);
        const unsigned target = phase * (unsigned)NWG;
        while (__hip_atomic_load(cnt, __ATOMIC_ACQUIRE,
                                 __HIP_MEMORY_SCOPE_AGENT) < target) {
            __builtin_amdgcn_s_sleep(2);
        }
    }
    __syncthreads();
    __threadfence();               // acquire side for all threads
}

__global__ void cvt_f32_bf16(const float* __restrict__ s,
                             __bf16* __restrict__ d, long n) {
    long i      = (long)blockIdx.x * blockDim.x + threadIdx.x;
    long stride = (long)gridDim.x * blockDim.x;
    for (; i < n; i += stride) d[i] = (__bf16)s[i];
}

__global__ void init_cnt(unsigned* c) {
    if (threadIdx.x == 0) *c = 0u;
}

__global__ __launch_bounds__(THREADS, 1)
void spiking_lstm_scan(const __bf16* __restrict__ xbf,   // [T,B,I] bf16
                       __bf16*                    y1bf,  // [T,B,H] bf16 (L0 out / L1 in)
                       const __bf16* __restrict__ wih,   // [L,G,I]  bf16 row-major
                       const __bf16* __restrict__ whh,   // [L,G,H]  bf16 row-major
                       const float*  __restrict__ bih,   // [L,G] fp32
                       const float*  __restrict__ bhh,   // [L,G] fp32
                       __bf16*                    hbuf,  // [2,B,H] bf16 double buffer
                       float*        __restrict__ out,   // y | h-states | c-states
                       unsigned*     __restrict__ gcnt)
{
    // Weight tiles for this WG, in WMMA-B lane order: lane ll owns 16
    // contiguous halves = W[j0+ (ll&15)][k0 + 8*(ll>>4) + {0..7,16..23}].  128 KB.
    __shared__ __align__(16) __bf16 wlds[2][4][KSTEPS][512];
    __shared__ float g4[4][BB][16];   // i/f/g/o spike tiles for this WG's cols
    __shared__ float cst[BB][16];     // resident cell state (WG-local columns)

    const int tid   = threadIdx.x;
    const int lane  = tid & 31;
    const int wv    = tid >> 5;       // 0..15
    const int mi    = wv & 3;         // row-tile of B=64
    const int gi    = wv >> 2;        // gate index 0..3 (i,f,g,o)
    const int m0    = mi * 16;
    const int ng    = blockIdx.x;     // hidden column group 0..31
    const int j0    = gi * H + ng * 16;   // gate-matrix column base in [0,2048)
    const int lrow  = lane & 15;
    const int lhalf = lane >> 4;

    unsigned phase = 0;

    // Zero this WG's columns of hbuf (both parities) so t=0 reads h=0.
    for (int e = tid; e < 2 * BB * 16; e += THREADS) {
        const int p = e / (BB * 16);
        const int r = (e >> 4) & (BB - 1);
        const int c = e & 15;
        hbuf[(size_t)p * BH + (size_t)r * H + ng * 16 + c] = (__bf16)0.0f;
    }
    gbar(gcnt, phase);

    for (int l = 0; l < L; ++l) {
        const __bf16* in = (l == 0) ? xbf : (const __bf16*)y1bf;
        const __bf16* Wi = wih + (size_t)l * G * I;
        const __bf16* Wh = whh + (size_t)l * G * H;
        const float   bj = bih[(size_t)l * G + j0 + lrow]
                         + bhh[(size_t)l * G + j0 + lrow];

        // ---- Stage this WG's weight tiles into LDS (once per layer). ----
        // 2 matrices x 4 gates x 16 ksteps x 512 halves = 8192 16B chunks.
        for (int c = tid; c < 2 * 4 * KSTEPS * 64; c += THREADS) {
            const int tile = c >> 6;          // 0..127
            const int wi_  = c & 63;          // chunk within tile
            const int ll   = wi_ >> 1;        // lane slot 0..31
            const int hh0  = (wi_ & 1) << 3;  // half offset 0 or 8
            const int mat  = tile >> 6;       // 0=Wih, 1=Whh
            const int gk   = tile & 63;
            const int gs   = gk >> 4;         // gate 0..3
            const int ks   = gk & 15;         // kstep 0..15
            const int j    = gs * H + ng * 16 + (ll & 15);
            const int k    = ks * 32 + ((ll >> 4) << 3) + (hh0 << 1);
            const __bf16* src = (mat == 0) ? (Wi + (size_t)j * I + k)
                                           : (Wh + (size_t)j * H + k);
            *(v8bf*)&wlds[mat][gs][ks][ll * 16 + hh0] = *(const v8bf*)src;
        }
        for (int e = tid; e < BB * 16; e += THREADS)
            cst[e >> 4][e & 15] = 0.0f;
        __syncthreads();

        // Per-lane A offsets (ISA 7.12.2 A layout): lanes 0-15 row=lane,
        // k {0..7,16..23}; lanes 16-31 same row, +8.
        const size_t aoff = (size_t)(m0 + lrow) * I + ((size_t)lhalf << 3);
        const size_t hoff = (size_t)(m0 + lrow) * H + ((size_t)lhalf << 3);
        const __bf16* bI = &wlds[0][gi][0][lane * 16];
        const __bf16* bH = &wlds[1][gi][0][lane * 16];

        for (int t = 0; t < T; ++t) {
            const int wp = t & 1;       // write parity
            const int rp = wp ^ 1;      // read parity

            // Stop LLVM from hoisting the t-invariant LDS weight loads out of
            // the loop into VGPRs (v1 spilled them to scratch).
            asm volatile("" ::: "memory");

            v8f acc;
            #pragma unroll
            for (int r = 0; r < 8; ++r) acc[r] = bj;   // bias folded into C

            // ---- input projection: x_t (16 x 512) x Wih tile ----
            {
                const __bf16* At = in + (size_t)t * BB * I + aoff;
                #pragma unroll
                for (int ks = 0; ks < KSTEPS; ++ks) {
                    const int k0 = ks * 32;
                    v8bf alo = *(const v8bf*)(At + k0);
                    v8bf ahi = *(const v8bf*)(At + k0 + 16);
                    v8bf blo = *(const v8bf*)(bI + (size_t)ks * 512);
                    v8bf bhi = *(const v8bf*)(bI + (size_t)ks * 512 + 8);
                    acc = __builtin_amdgcn_wmma_f32_16x16x32_bf16(
                        false, cat8(alo, ahi), false, cat8(blo, bhi),
                        (short)0, acc, false, false);
                }
            }
            // ---- recurrent projection: h (16 x 512) x Whh tile ----
            {
                const __bf16* At = hbuf + (size_t)rp * BH + hoff;
                #pragma unroll
                for (int ks = 0; ks < KSTEPS; ++ks) {
                    const int k0 = ks * 32;
                    v8bf alo = *(const v8bf*)(At + k0);
                    v8bf ahi = *(const v8bf*)(At + k0 + 16);
                    v8bf blo = *(const v8bf*)(bH + (size_t)ks * 512);
                    v8bf bhi = *(const v8bf*)(bH + (size_t)ks * 512 + 8);
                    acc = __builtin_amdgcn_wmma_f32_16x16x32_bf16(
                        false, cat8(alo, ahi), false, cat8(blo, bhi),
                        (short)0, acc, false, false);
                }
            }

            // Heaviside spike; C-layout: VGPR r -> M = r + 8*lhalf, N = lrow.
            #pragma unroll
            for (int r = 0; r < 8; ++r) {
                g4[gi][m0 + r + (lhalf << 3)][lrow] =
                    (acc[r] >= 0.0f) ? 1.0f : 0.0f;
            }
            __syncthreads();

            // Cell/hidden update for this WG's 64x16 hidden-column block.
            for (int e = tid; e < BB * 16; e += THREADS) {
                const int row = e >> 4, col = e & 15;
                const float i_ = g4[0][row][col];
                const float f_ = g4[1][row][col];
                const float g_ = g4[2][row][col];
                const float o_ = g4[3][row][col];
                float c = cst[row][col];
                c = c * f_ + i_ * g_;
                const float h = c * o_;
                cst[row][col] = c;
                const int gcol = ng * 16 + col;
                hbuf[(size_t)wp * BH + (size_t)row * H + gcol] = (__bf16)h;
                if (l == 0) {
                    y1bf[((size_t)t * BB + row) * H + gcol] = (__bf16)h;
                } else {
                    out[((size_t)t * BB + row) * H + gcol] = h;
                }
                if (t == T - 1) {
                    out[YELEMS + ((size_t)l * BB + row) * H + gcol] = h;
                    out[YELEMS + (size_t)L * BH
                               + ((size_t)l * BB + row) * H + gcol] = c;
                }
            }
            gbar(gcnt, phase);   // publish h(t); orders reads(t) < writes(t+1)
        }

        if (l == 0) {
            // Reset h double buffer for layer 1.
            for (int e = tid; e < 2 * BB * 16; e += THREADS) {
                const int p = e / (BB * 16);
                const int r = (e >> 4) & (BB - 1);
                const int c = e & 15;
                hbuf[(size_t)p * BH + (size_t)r * H + ng * 16 + c] = (__bf16)0.0f;
            }
            gbar(gcnt, phase);
        }
    }
}

extern "C" void kernel_launch(void* const* d_in, const int* in_sizes, int n_in,
                              void* d_out, int out_size, void* d_ws, size_t ws_size,
                              hipStream_t stream) {
    (void)in_sizes; (void)n_in; (void)out_size; (void)ws_size;
    const float* x    = (const float*)d_in[0];   // [T,B,I]
    const float* Wih  = (const float*)d_in[1];   // [L,4H,I]
    const float* Whh  = (const float*)d_in[2];   // [L,4H,H]
    const float* bih_ = (const float*)d_in[3];   // [L,4H]
    const float* bhh_ = (const float*)d_in[4];   // [L,4H]

    char*  ws  = (char*)d_ws;
    size_t off = 0;
    unsigned* cnt  = (unsigned*)(ws + off); off += 256;
    __bf16*   xbf  = (__bf16*)(ws + off);   off += (size_t)T * BB * I * 2;
    __bf16*   y1bf = (__bf16*)(ws + off);   off += (size_t)T * BB * H * 2;
    __bf16*   wihb = (__bf16*)(ws + off);   off += (size_t)L * G * I * 2;
    __bf16*   whhb = (__bf16*)(ws + off);   off += (size_t)L * G * H * 2;
    __bf16*   hbuf = (__bf16*)(ws + off);   off += (size_t)2 * BB * H * 2;

    cvt_f32_bf16<<<2048, 256, 0, stream>>>(x,   xbf,  (long)T * BB * I);
    cvt_f32_bf16<<<512,  256, 0, stream>>>(Wih, wihb, (long)L * G * I);
    cvt_f32_bf16<<<512,  256, 0, stream>>>(Whh, whhb, (long)L * G * H);
    init_cnt<<<1, 32, 0, stream>>>(cnt);

    spiking_lstm_scan<<<NWG, THREADS, 0, stream>>>(
        xbf, y1bf, wihb, whhb, bih_, bhh_, hbuf, (float*)d_out, cnt);
}
